// GNN_20839181320253
// MI455X (gfx1250) — matrix operand order repaired and checked by reading, not compiled, not run on hardware
//
#include <hip/hip_runtime.h>
#include <hip/hip_bf16.h>
#include <stdint.h>

// ---------------------------------------------------------------------------
// Types for CDNA5 WMMA (wave32, v_wmma_f32_16x16x32_bf16)
// ---------------------------------------------------------------------------
typedef __bf16 bf16_t;
typedef __attribute__((ext_vector_type(16))) __bf16 v16bf;
typedef __attribute__((ext_vector_type(8)))  __bf16 v8bf;
typedef __attribute__((ext_vector_type(8)))  float  v8f;

static __device__ __forceinline__ bf16_t f2bf(float f) {
  union { float f; unsigned u; } v; v.f = f;
  unsigned r = v.u + 0x7FFFu + ((v.u >> 16) & 1u);   // round-to-nearest-even
  unsigned short hs = (unsigned short)(r >> 16);
  return __builtin_bit_cast(bf16_t, hs);
}

// ---------------------------------------------------------------------------
// Problem constants (from reference)
// ---------------------------------------------------------------------------
#define NB   30
#define NPG  1000
#define EPG  2500
#define NN   (NB * NPG)     // 30000 nodes per batch
#define EE   (NB * EPG)     // 75000 edges per batch
#define SLOPE 0.01f
#define BN_EPS 1e-5f

static inline int pad32(int x) { return (x + 31) & ~31; }
static inline int pad16(int x) { return (x + 15) & ~15; }

// ---------------------------------------------------------------------------
// Kernels
// ---------------------------------------------------------------------------

// Pack f32 W [K x Nout] into WMMA B-fragment order so a lane's 16 bf16 are
// contiguous:
//   out[(((tn*nk)+tk)*32 + lane)*16 + i] =
//       W[(tk*32 + 16*(lane>>4) + i) * Nout + tn*16 + (lane&15)]   (0 if OOB)
__global__ void k_pack_wfrag(const float* __restrict__ W, int K, int Nout, int Kp,
                             bf16_t* __restrict__ out) {
  int nk = Kp >> 5;
  int tilesn = (Nout + 15) >> 4;
  size_t tot = (size_t)tilesn * nk * 32 * 16;
  size_t i = (size_t)blockIdx.x * blockDim.x + threadIdx.x;
  if (i >= tot) return;
  int e    = (int)(i & 15);
  int lane = (int)((i >> 4) & 31);
  int tkn  = (int)(i >> 9);
  int tk   = tkn % nk;
  int tn   = tkn / nk;
  int r  = tk * 32 + 16 * (lane >> 4) + e;
  int cc = tn * 16 + (lane & 15);
  float v = (r < K && cc < Nout) ? W[(size_t)r * Nout + cc] : 0.0f;
  out[i] = f2bf(v);
}

// out[MxNout] = act( A[pad16(M)xKp](bf16) * Wfrag + bias ), f32 accum via WMMA.
// A rows are zero-padded to >= pad16(M) so loads need no guards.
// 4 waves/block; wave w owns rows [(bx*4+w)*16, +16), cols [by*16, +16).
__global__ void k_gemm_bf16(const bf16_t* __restrict__ A,
                            const bf16_t* __restrict__ Wf,
                            const float*  __restrict__ bias,
                            float*        __restrict__ out,
                            int M, int Kp, int Nout, int act) {
  const int lane = threadIdx.x & 31;
  const int wv   = threadIdx.x >> 5;
  const int m0   = (blockIdx.x * 4 + wv) * 16;
  const int n0   = blockIdx.y * 16;
  if (m0 >= M) return;                 // wave-uniform: EXEC stays all-1 for WMMA

  const int half = lane >> 4;
  const int l15  = lane & 15;
  const int nk   = Kp >> 5;

  const bf16_t* arow  = A + (size_t)(m0 + l15) * Kp + 16 * half;
  const bf16_t* wfrag = Wf + ((size_t)blockIdx.y * nk * 32 + lane) * 16;

  v8f acc = {};
  for (int t = 0; t < nk; ++t) {
    v8bf lo = *(const v8bf*)(arow);
    v8bf hi = *(const v8bf*)(arow + 8);
    v16bf a = __builtin_shufflevector(lo, hi,
               0,1,2,3,4,5,6,7,8,9,10,11,12,13,14,15);
    v16bf b = *(const v16bf*)(wfrag);
    acc = __builtin_amdgcn_wmma_f32_16x16x32_bf16(false, a, false, b,
                                                  (short)0, acc, false, false);
    arow  += 32;
    wfrag += 32 * 16;
  }

  const int col = n0 + l15;
  if (col < Nout) {
    const float bv = bias[col];
    #pragma unroll
    for (int r = 0; r < 8; ++r) {
      int row = m0 + r + 8 * half;     // C/D layout: lanes 16-31 hold M = r+8
      if (row < M) {
        float v = acc[r] + bv;
        if (act) v = (v >= 0.0f) ? v : SLOPE * v;
        out[(size_t)row * Nout + col] = v;
      }
    }
  }
}

// Per-column sum & sum-of-squares (one block per column).
__global__ void k_colstats(const float* __restrict__ X, int M, int Ncol,
                           float* __restrict__ sums, float* __restrict__ sqs) {
  __shared__ float s1[256], s2[256];
  int col = blockIdx.x;
  float a = 0.f, b = 0.f;
  for (int r = threadIdx.x; r < M; r += blockDim.x) {
    float v = X[(size_t)r * Ncol + col];
    a += v; b += v * v;
  }
  s1[threadIdx.x] = a; s2[threadIdx.x] = b;
  __syncthreads();
  for (int s = blockDim.x >> 1; s > 0; s >>= 1) {
    if (threadIdx.x < (unsigned)s) {
      s1[threadIdx.x] += s1[threadIdx.x + s];
      s2[threadIdx.x] += s2[threadIdx.x + s];
    }
    __syncthreads();
  }
  if (threadIdx.x == 0) { sums[col] = s1[0]; sqs[col] = s2[0]; }
}

// BatchNorm (training, biased var) + convert to zero-padded bf16 [Mp x Kp].
__global__ void k_bn_to_bf16(const float* __restrict__ X, int M, int Mp,
                             int Ncol, int Kp,
                             const float* __restrict__ sums, const float* __restrict__ sqs,
                             const float* __restrict__ g, const float* __restrict__ be,
                             bf16_t* __restrict__ out) {
  size_t i = (size_t)blockIdx.x * blockDim.x + threadIdx.x;
  size_t tot = (size_t)Mp * Kp;
  if (i >= tot) return;
  int col = (int)(i % Kp);
  int row = (int)(i / Kp);
  float v = 0.0f;
  if (col < Ncol && row < M) {
    float invM  = 1.0f / (float)M;
    float mean  = sums[col] * invM;
    float var   = sqs[col] * invM - mean * mean;
    float scale = g[col] * rsqrtf(var + BN_EPS);
    v = (X[(size_t)row * Ncol + col] - mean) * scale + be[col];
  }
  out[i] = f2bf(v);
}

// [x[row], x[col], ea] -> bf16 [Ep x Kp] (rows >= E zero)
__global__ void k_concat_edge(const float* __restrict__ x, int Fx,
                              const float* __restrict__ ea, int Fe,
                              const int* __restrict__ row, const int* __restrict__ col,
                              int E, int Ep, int Kp, bf16_t* __restrict__ out) {
  size_t i = (size_t)blockIdx.x * blockDim.x + threadIdx.x;
  size_t tot = (size_t)Ep * Kp;
  if (i >= tot) return;
  int e = (int)(i / Kp), k = (int)(i % Kp);
  float v = 0.0f;
  if (e < E) {
    if (k < Fx)               v = x[(size_t)row[e] * Fx + k];
    else if (k < 2 * Fx)      v = x[(size_t)col[e] * Fx + (k - Fx)];
    else if (k < 2 * Fx + Fe) v = ea[(size_t)e * Fe + (k - 2 * Fx)];
  }
  out[i] = f2bf(v);
}

// [x[row], ea] -> bf16 [Ep x Kp]
__global__ void k_concat_gather(const float* __restrict__ x, int Fx,
                                const float* __restrict__ ea, int Fe,
                                const int* __restrict__ row,
                                int E, int Ep, int Kp, bf16_t* __restrict__ out) {
  size_t i = (size_t)blockIdx.x * blockDim.x + threadIdx.x;
  size_t tot = (size_t)Ep * Kp;
  if (i >= tot) return;
  int e = (int)(i / Kp), k = (int)(i % Kp);
  float v = 0.0f;
  if (e < E) {
    if (k < Fx)           v = x[(size_t)row[e] * Fx + k];
    else if (k < Fx + Fe) v = ea[(size_t)e * Fe + (k - Fx)];
  }
  out[i] = f2bf(v);
}

// [a, b] (row-aligned) -> bf16 [Mp x Kp]
__global__ void k_concat2(const float* __restrict__ a, int Fa,
                          const float* __restrict__ b, int Fb,
                          int M, int Mp, int Kp, bf16_t* __restrict__ out) {
  size_t i = (size_t)blockIdx.x * blockDim.x + threadIdx.x;
  size_t tot = (size_t)Mp * Kp;
  if (i >= tot) return;
  int m = (int)(i / Kp), k = (int)(i % Kp);
  float v = 0.0f;
  if (m < M) {
    if (k < Fa)           v = a[(size_t)m * Fa + k];
    else if (k < Fa + Fb) v = b[(size_t)m * Fb + (k - Fa)];
  }
  out[i] = f2bf(v);
}

__global__ void k_zero(float* __restrict__ p, size_t n) {
  size_t i = (size_t)blockIdx.x * blockDim.x + threadIdx.x;
  if (i < n) p[i] = 0.0f;
}

// dst[seg[m]*F + f] += src[m*F + f]
__global__ void k_segsum(const float* __restrict__ src, int F,
                         const int* __restrict__ seg, int M,
                         float* __restrict__ dst) {
  size_t i = (size_t)blockIdx.x * blockDim.x + threadIdx.x;
  size_t tot = (size_t)M * F;
  if (i >= tot) return;
  int m = (int)(i / F), f = (int)(i % F);
  atomicAdd(&dst[(size_t)seg[m] * F + f], src[i]);
}

// nu graph builders
__global__ void k_build_nux(const float* __restrict__ u1, const float* __restrict__ u2,
                            const float* __restrict__ um, float* __restrict__ nx) {
  int i = blockIdx.x * blockDim.x + threadIdx.x;          // 90*64
  if (i >= 3 * NB * 64) return;
  int nrow = i >> 6, f = i & 63;
  int b = nrow / 3, j = nrow % 3;
  const float* u = (j == 0) ? u1 : (j == 1 ? u2 : um);
  nx[i] = u[b * 64 + f];
}
__global__ void k_build_nuei(int* __restrict__ row, int* __restrict__ col) {
  int i = blockIdx.x * blockDim.x + threadIdx.x;          // 120
  if (i >= 4 * NB) return;
  const int tr[4] = {0, 2, 1, 2};
  const int tc[4] = {2, 0, 2, 1};
  int b = i >> 2, t = i & 3;
  row[i] = tr[t] + 3 * b;
  col[i] = tc[t] + 3 * b;
}
__global__ void k_build_nuea(const float* __restrict__ up1, const float* __restrict__ up2,
                             const float* __restrict__ upm, float* __restrict__ ea) {
  int i = blockIdx.x * blockDim.x + threadIdx.x;          // 120*32
  if (i >= 4 * NB * 32) return;
  int e = i >> 5, b = e >> 2, t = e & 3;
  ea[i] = ((t < 2) ? up1[b] : up2[b]) / upm[b];
}
__global__ void k_build_nubatch(int* __restrict__ nb) {
  int i = blockIdx.x * blockDim.x + threadIdx.x;          // 90
  if (i < 3 * NB) nb[i] = i / 3;
}

// ---------------------------------------------------------------------------
// Host orchestration
// ---------------------------------------------------------------------------
struct Mlp {
  const float* W[4]; const float* b[4]; const float* g[3]; const float* be[3];
  bf16_t* Wp[4];
  int dims[5];
  int nlin;
};

struct Ctx {
  hipStream_t s;
  char*  base;
  size_t off;
  size_t cap;
  bf16_t *bfIn, *bfAct;
  float  *hbuf, *mbuf, *agg, *poolx, *poolg, *sums, *sqs;
};

static void* wsalloc(Ctx& c, size_t bytes) {
  void* p = c.base + c.off;
  c.off = (c.off + bytes + 255) & ~(size_t)255;
  return p;
}
static inline unsigned nb256(size_t n) { return (unsigned)((n + 255) / 256); }

// Param leaf order: jax pytree (dict keys sorted). Per MLP: Ws[], bes[], bs[], gs[].
static int mk_mlp(Mlp& m, void* const* din, int base, const int* dims, int nd) {
  int n = nd - 1, k = nd - 2, idx = base;
  for (int i = 0; i < n; ++i) m.W[i]  = (const float*)din[idx++];
  for (int i = 0; i < k; ++i) m.be[i] = (const float*)din[idx++];
  for (int i = 0; i < n; ++i) m.b[i]  = (const float*)din[idx++];
  for (int i = 0; i < k; ++i) m.g[i]  = (const float*)din[idx++];
  for (int i = 0; i < nd; ++i) m.dims[i] = dims[i];
  m.nlin = n;
  return idx;
}

static void pack_mlp(Ctx& c, Mlp& m) {
  for (int i = 0; i < m.nlin; ++i) {
    int K = m.dims[i], N = m.dims[i + 1], Kp = pad32(K);
    size_t tot = (size_t)((N + 15) / 16) * (Kp >> 5) * 32 * 16;
    m.Wp[i] = (bf16_t*)wsalloc(c, tot * sizeof(bf16_t));
    k_pack_wfrag<<<nb256(tot), 256, 0, c.s>>>(m.W[i], K, N, Kp, m.Wp[i]);
  }
}

static void gemm(Ctx& c, const bf16_t* A, int Kp, const bf16_t* W, const float* bias,
                 float* out, int M, int No, int act) {
  dim3 g((M + 63) / 64, (No + 15) / 16), blk(128);
  k_gemm_bf16<<<g, blk, 0, c.s>>>(A, W, bias, out, M, Kp, No, act);
}

static void run_mlp(Ctx& c, Mlp& m, const bf16_t* A0, int M, float* out) {
  const bf16_t* A = A0;
  int Mp = pad16(M);
  for (int i = 0; i < m.nlin - 1; ++i) {
    int No = m.dims[i + 1];
    gemm(c, A, pad32(m.dims[i]), m.Wp[i], m.b[i], c.hbuf, M, No, /*act=*/1);
    k_colstats<<<No, 256, 0, c.s>>>(c.hbuf, M, No, c.sums, c.sqs);
    size_t tot = (size_t)Mp * pad32(No);
    k_bn_to_bf16<<<nb256(tot), 256, 0, c.s>>>(c.hbuf, M, Mp, No, pad32(No),
                                              c.sums, c.sqs, m.g[i], m.be[i], c.bfAct);
    A = c.bfAct;
  }
  int L = m.nlin - 1;
  gemm(c, A, pad32(m.dims[L]), m.Wp[L], m.b[L], out, M, m.dims[L + 1], /*act=*/0);
}

struct MetaP { Mlp edge, node1, node2, glob1, glob2; };

// PyG MetaLayer step. `u` is never read inside _meta, so glob path only when do_glob.
static void run_meta(Ctx& c, MetaP& P, float* x, int Nn, int Fx,
                     const int* row, const int* col, int Ee, float* ea, int Fe,
                     const int* batch, float* u_out, bool do_glob) {
  int Hn = P.node1.dims[P.node1.nlin];   // node1 output width
  int Ep = pad16(Ee), Np = pad16(Nn);
  // edge MLP: ea = mlp_e([x[row], x[col], ea])
  {
    int Kp = pad32(2 * Fx + Fe);
    size_t tot = (size_t)Ep * Kp;
    k_concat_edge<<<nb256(tot), 256, 0, c.s>>>(x, Fx, ea, Fe, row, col, Ee, Ep, Kp, c.bfIn);
  }
  run_mlp(c, P.edge, c.bfIn, Ee, ea);
  // node1 MLP: m = mlp_n1([x[row], ea])
  {
    int Kp = pad32(Fx + Fe);
    size_t tot = (size_t)Ep * Kp;
    k_concat_gather<<<nb256(tot), 256, 0, c.s>>>(x, Fx, ea, Fe, row, Ee, Ep, Kp, c.bfIn);
  }
  run_mlp(c, P.node1, c.bfIn, Ee, c.mbuf);
  // agg = segment_sum(m, col, Nn)
  {
    size_t an = (size_t)Nn * Hn;
    k_zero<<<nb256(an), 256, 0, c.s>>>(c.agg, an);
    size_t tot = (size_t)Ee * Hn;
    k_segsum<<<nb256(tot), 256, 0, c.s>>>(c.mbuf, Hn, col, Ee, c.agg);
  }
  // node2 MLP: x = mlp_n2([x, agg])
  {
    int Kp = pad32(Fx + Hn);
    size_t tot = (size_t)Np * Kp;
    k_concat2<<<nb256(tot), 256, 0, c.s>>>(x, Fx, c.agg, Hn, Nn, Np, Kp, c.bfIn);
  }
  run_mlp(c, P.node2, c.bfIn, Nn, x);
  if (!do_glob) return;
  // glob path (only needed on last iteration)
  int Hg = P.glob1.dims[P.glob1.nlin];
  {
    int Kp = pad32(Fx + Fe);
    size_t tot = (size_t)Ep * Kp;
    k_concat_gather<<<nb256(tot), 256, 0, c.s>>>(x, Fx, ea, Fe, row, Ee, Ep, Kp, c.bfIn);
  }
  run_mlp(c, P.glob1, c.bfIn, Ee, c.mbuf);
  {
    size_t an = (size_t)Nn * Hg;
    k_zero<<<nb256(an), 256, 0, c.s>>>(c.agg, an);
    size_t tot = (size_t)Ee * Hg;
    k_segsum<<<nb256(tot), 256, 0, c.s>>>(c.mbuf, Hg, col, Ee, c.agg);
  }
  {
    size_t t1 = (size_t)NB * Fx, t2 = (size_t)NB * Hg;
    k_zero<<<nb256(t1), 256, 0, c.s>>>(c.poolx, t1);
    k_zero<<<nb256(t2), 256, 0, c.s>>>(c.poolg, t2);
    size_t tx = (size_t)Nn * Fx, tg = (size_t)Nn * Hg;
    k_segsum<<<nb256(tx), 256, 0, c.s>>>(x, Fx, batch, Nn, c.poolx);
    k_segsum<<<nb256(tg), 256, 0, c.s>>>(c.agg, Hg, batch, Nn, c.poolg);
  }
  {
    int Kp = pad32(Fx + Hg);
    size_t tot = (size_t)pad16(NB) * Kp;
    k_concat2<<<nb256(tot), 256, 0, c.s>>>(c.poolx, Fx, c.poolg, Hg, NB, pad16(NB), Kp, c.bfIn);
  }
  run_mlp(c, P.glob2, c.bfIn, NB, u_out);
}

// d_in leaf indices (jax pytree order: last, m1.{edge,glob1,glob2,node1,node2}, m3.{...}, data)
enum {
  P_LAST = 0,
  P_M1_EDGE = 6, P_M1_GLOB1 = 12, P_M1_GLOB2 = 18, P_M1_NODE1 = 32, P_M1_NODE2 = 38,
  P_M3_EDGE = 44, P_M3_GLOB1 = 50, P_M3_GLOB2 = 56, P_M3_NODE1 = 70, P_M3_NODE2 = 76,
  IN_X1 = 82, IN_EI1 = 83, IN_EA1 = 84, IN_U1 = 85, IN_B1 = 86,
  IN_X2 = 87, IN_EI2 = 88, IN_EA2 = 89, IN_U2 = 90, IN_B2 = 91,
  IN_XM = 92, IN_EIM = 93, IN_EAM = 94, IN_UM = 95, IN_BM = 96,
  IN_TEMP = 97
};

extern "C" void kernel_launch(void* const* d_in, const int* in_sizes, int n_in,
                              void* d_out, int out_size, void* d_ws, size_t ws_size,
                              hipStream_t stream) {
  (void)in_sizes; (void)n_in; (void)out_size;
  Ctx c; c.s = stream; c.base = (char*)d_ws; c.off = 0; c.cap = ws_size;

  // ---- build param structs ----
  static const int d_last[]  = {65, 64, 15};
  static const int d1_edge[] = {20, 64, 6};
  static const int d1_g1[]   = {13, 64, 64};
  static const int d1_g2[]   = {71, 64, 64, 64, 64};
  static const int d1_n1[]   = {13, 64, 64};
  static const int d1_n2[]   = {71, 64, 7};
  static const int d3_edge[] = {160, 128, 32};
  static const int d3_g1[]   = {96, 128, 128};
  static const int d3_g2[]   = {192, 128, 128, 128, 64};
  static const int d3_n1[]   = {96, 128, 128};
  static const int d3_n2[]   = {192, 128, 64};

  Mlp last; MetaP m1, m3;
  mk_mlp(last,     d_in, P_LAST,     d_last,  3);
  mk_mlp(m1.edge,  d_in, P_M1_EDGE,  d1_edge, 3);
  mk_mlp(m1.glob1, d_in, P_M1_GLOB1, d1_g1,   3);
  mk_mlp(m1.glob2, d_in, P_M1_GLOB2, d1_g2,   5);
  mk_mlp(m1.node1, d_in, P_M1_NODE1, d1_n1,   3);
  mk_mlp(m1.node2, d_in, P_M1_NODE2, d1_n2,   3);
  mk_mlp(m3.edge,  d_in, P_M3_EDGE,  d3_edge, 3);
  mk_mlp(m3.glob1, d_in, P_M3_GLOB1, d3_g1,   3);
  mk_mlp(m3.glob2, d_in, P_M3_GLOB2, d3_g2,   5);
  mk_mlp(m3.node1, d_in, P_M3_NODE1, d3_n1,   3);
  mk_mlp(m3.node2, d_in, P_M3_NODE2, d3_n2,   3);

  // ---- workspace carve (padded rows: +16) ----
  c.bfIn  = (bf16_t*)wsalloc(c, (size_t)(EE + 16) * 64 * sizeof(bf16_t));
  c.bfAct = (bf16_t*)wsalloc(c, (size_t)(EE + 16) * 64 * sizeof(bf16_t));
  c.hbuf  = (float*) wsalloc(c, (size_t)EE * 64 * sizeof(float));
  c.mbuf  = (float*) wsalloc(c, (size_t)EE * 64 * sizeof(float));
  c.agg   = (float*) wsalloc(c, (size_t)NN * 64 * sizeof(float));
  c.poolx = (float*) wsalloc(c, (size_t)NB * 128 * sizeof(float));
  c.poolg = (float*) wsalloc(c, (size_t)NB * 128 * sizeof(float));
  c.sums  = (float*) wsalloc(c, 256 * sizeof(float));
  c.sqs   = (float*) wsalloc(c, 256 * sizeof(float));

  float* x_cur[3];  float* ea_cur[3];  float* u_out[3];
  for (int g = 0; g < 3; ++g) {
    x_cur[g]  = (float*)wsalloc(c, (size_t)NN * 7 * sizeof(float));
    ea_cur[g] = (float*)wsalloc(c, (size_t)EE * 6 * sizeof(float));
    u_out[g]  = (float*)wsalloc(c, (size_t)NB * 64 * sizeof(float));
  }
  float* nu_x  = (float*)wsalloc(c, (size_t)3 * NB * 64 * sizeof(float));
  float* nu_ea = (float*)wsalloc(c, (size_t)4 * NB * 32 * sizeof(float));
  float* nu_u  = (float*)wsalloc(c, (size_t)NB * 64 * sizeof(float));
  int* nu_row   = (int*)wsalloc(c, 4 * NB * sizeof(int));
  int* nu_col   = (int*)wsalloc(c, 4 * NB * sizeof(int));
  int* nu_batch = (int*)wsalloc(c, 3 * NB * sizeof(int));

  // ---- pack all weights to WMMA B-fragment layout once ----
  pack_mlp(c, last);
  pack_mlp(c, m1.edge); pack_mlp(c, m1.glob1); pack_mlp(c, m1.glob2);
  pack_mlp(c, m1.node1); pack_mlp(c, m1.node2);
  pack_mlp(c, m3.edge); pack_mlp(c, m3.glob1); pack_mlp(c, m3.glob2);
  pack_mlp(c, m3.node1); pack_mlp(c, m3.node2);

  // ---- copy mutable graph state into ws (inputs must not be modified) ----
  const int xi[3]  = {IN_X1, IN_X2, IN_XM};
  const int eai[3] = {IN_EA1, IN_EA2, IN_EAM};
  const int eii[3] = {IN_EI1, IN_EI2, IN_EIM};
  const int bti[3] = {IN_B1, IN_B2, IN_BM};
  for (int g = 0; g < 3; ++g) {
    hipMemcpyAsync(x_cur[g],  d_in[xi[g]],  (size_t)NN * 7 * sizeof(float),
                   hipMemcpyDeviceToDevice, stream);
    hipMemcpyAsync(ea_cur[g], d_in[eai[g]], (size_t)EE * 6 * sizeof(float),
                   hipMemcpyDeviceToDevice, stream);
  }

  // ---- phase 1: 3 MP rounds x 3 graph batches; glob path only on last round ----
  for (int it = 0; it < 3; ++it) {
    for (int g = 0; g < 3; ++g) {
      const int* ei  = (const int*)d_in[eii[g]];
      const int* row = ei;            // ei[0]
      const int* col = ei + EE;       // ei[1]
      const int* bt  = (const int*)d_in[bti[g]];
      run_meta(c, m1, x_cur[g], NN, 7, row, col, EE, ea_cur[g], 6,
               bt, u_out[g], /*do_glob=*/(it == 2));
    }
  }

  // ---- build nu graph ----
  k_build_nux<<<nb256(3 * NB * 64), 256, 0, stream>>>(u_out[0], u_out[1], u_out[2], nu_x);
  k_build_nuei<<<1, 128, 0, stream>>>(nu_row, nu_col);
  k_build_nuea<<<nb256(4 * NB * 32), 256, 0, stream>>>(
      (const float*)d_in[IN_U1], (const float*)d_in[IN_U2], (const float*)d_in[IN_UM], nu_ea);
  k_build_nubatch<<<1, 96, 0, stream>>>(nu_batch);

  // ---- phase 2: 6 MP rounds on nu graph ----
  for (int it = 0; it < 6; ++it) {
    run_meta(c, m3, nu_x, 3 * NB, 64, nu_row, nu_col, 4 * NB, nu_ea, 32,
             nu_batch, nu_u, /*do_glob=*/(it == 5));
  }

  // ---- head: concat([Temperature, nu_u]) -> last MLP -> [30,15] ----
  {
    int Kp = pad32(65);
    size_t tot = (size_t)pad16(NB) * Kp;
    k_concat2<<<nb256(tot), 256, 0, stream>>>((const float*)d_in[IN_TEMP], 1,
                                              nu_u, 64, NB, pad16(NB), Kp, c.bfIn);
  }
  run_mlp(c, last, c.bfIn, NB, (float*)d_out);
}
